// NegSupConLoss_46514495815718
// MI455X (gfx1250) — matrix operand order repaired and checked by reading, not compiled, not run on hardware
//
#include <hip/hip_runtime.h>
#include <hip/hip_bf16.h>

typedef __attribute__((ext_vector_type(8)))  __bf16 v8bf;
typedef __attribute__((ext_vector_type(16))) __bf16 v16bf;
typedef __attribute__((ext_vector_type(8)))  float  v8f;
typedef __attribute__((ext_vector_type(4)))  int    v4i;

#define B_ROWS   4096            // batch B (power of two -> partner = i ^ B_ROWS)
#define NROWS    8192            // N = 2B
#define DDIM     256             // feature dim
#define NTILES   (NROWS / 16)    // 512 column tiles
#define TEMP_INV 2.0f            // 1 / temperature
#define NSPLIT   8               // column splits across blocks
#define TILES_PER_BLOCK (NTILES / NSPLIT)   // 64
#define LDS_STRIDE 280           // bf16 elements per tile column (256 + 24 pad)

// -------- gfx1250 async global->LDS path (guarded; plain copy fallback) ----
#ifndef USE_ASYNC_LDS
#if defined(__has_builtin)
#if __has_builtin(__builtin_amdgcn_global_load_async_to_lds_b128)
#define USE_ASYNC_LDS 1
#endif
#endif
#endif
#ifndef USE_ASYNC_LDS
#define USE_ASYNC_LDS 0
#endif

#if USE_ASYNC_LDS
typedef __attribute__((address_space(1))) v4i glob_v4i;   // global int4
typedef __attribute__((address_space(3))) v4i lds_v4i;    // LDS int4
#endif

__device__ __forceinline__ void async_copy16B(const __bf16* g, __bf16* l) {
#if USE_ASYNC_LDS
  __builtin_amdgcn_global_load_async_to_lds_b128(
      (glob_v4i*)g, (lds_v4i*)l, 0, 0);
#else
  *(v8bf*)l = *(const v8bf*)g;
#endif
}

__device__ __forceinline__ void async_wait_all() {
#if USE_ASYNC_LDS
#if defined(__has_builtin) && __has_builtin(__builtin_amdgcn_s_wait_asynccnt)
  __builtin_amdgcn_s_wait_asynccnt(0);
#else
  asm volatile("s_wait_asynccnt 0x0" ::: "memory");
#endif
#endif
}

// ---------------------------------------------------------------------------
// Phase 1: row L2-normalize z = [z_i; z_j], convert to bf16, store to ws.
// ---------------------------------------------------------------------------
__global__ __launch_bounds__(256) void nscl_normalize(const float* __restrict__ zi,
                                                      const float* __restrict__ zj,
                                                      __bf16* __restrict__ zn) {
  const int wave = threadIdx.x >> 5;
  const int lane = threadIdx.x & 31;
  const int row  = blockIdx.x * 8 + wave;
  if (row >= NROWS) return;

  const float* src = (row < B_ROWS) ? (zi + (size_t)row * DDIM)
                                    : (zj + (size_t)(row - B_ROWS) * DDIM);
  const float4* s4 = (const float4*)(src + lane * 8);
  float4 a = s4[0];
  float4 b = s4[1];

  float ss = a.x*a.x + a.y*a.y + a.z*a.z + a.w*a.w
           + b.x*b.x + b.y*b.y + b.z*b.z + b.w*b.w;
#pragma unroll
  for (int m = 16; m >= 1; m >>= 1) ss += __shfl_xor(ss, m, 32);

  const float inv = 1.0f / fmaxf(sqrtf(ss), 1e-8f);

  v8bf o;
  o[0] = (__bf16)(a.x * inv); o[1] = (__bf16)(a.y * inv);
  o[2] = (__bf16)(a.z * inv); o[3] = (__bf16)(a.w * inv);
  o[4] = (__bf16)(b.x * inv); o[5] = (__bf16)(b.y * inv);
  o[6] = (__bf16)(b.z * inv); o[7] = (__bf16)(b.w * inv);
  *(v8bf*)(zn + (size_t)row * DDIM + lane * 8) = o;
}

// ---------------------------------------------------------------------------
// Phase 2: block = 8 waves x 32 rows = 256 rows, 64 column tiles (1/8 of N).
// B tiles staged in LDS (double-buffered, async-copy when available) and
// shared by all 8 waves; each B fragment feeds 2 WMMAs (2 row stripes).
// Per-split partial sums/pos go to workspace.
// ---------------------------------------------------------------------------
__global__ __launch_bounds__(256) void nscl_simloss(const __bf16* __restrict__ zn,
                                                    const int* __restrict__ lab,
                                                    float* __restrict__ partSum,
                                                    float* __restrict__ partPos) {
  __shared__ __bf16 Btile[2][16 * LDS_STRIDE];   // 2 x ~8.75 KB

  const int wave = threadIdx.x >> 5;
  const int lane = threadIdx.x & 31;
  const int half = lane >> 4;
  const int l16  = lane & 15;

  const int rg  = blockIdx.x >> 3;               // 32 row-groups
  const int cs  = blockIdx.x & (NSPLIT - 1);     // 8 column splits
  const int iw0 = rg * 256 + wave * 32;          // this wave's 32 rows
  const int n0base = cs * TILES_PER_BLOCK * 16;  // first column of split

  // ---- preload A: 2 stripes x 8 K-step fragments (ISA 16-bit A layout) ----
  v16bf Af[2][8];
#pragma unroll
  for (int s = 0; s < 2; ++s) {
    const __bf16* rowp = zn + (size_t)(iw0 + s * 16 + l16) * DDIM;
    const int o = half ? 8 : 0;
#pragma unroll
    for (int k = 0; k < 8; ++k) {
      v8bf lo = *(const v8bf*)(rowp + k * 32 + o);
      v8bf hi = *(const v8bf*)(rowp + k * 32 + 16 + o);
      Af[s][k] = __builtin_shufflevector(lo, hi,
                   0, 1, 2, 3, 4, 5, 6, 7, 8, 9, 10, 11, 12, 13, 14, 15);
    }
  }

  int labI[2][8];
#pragma unroll
  for (int s = 0; s < 2; ++s)
#pragma unroll
    for (int r = 0; r < 8; ++r)
      labI[s][r] = lab[(iw0 + s * 16 + r + (half ? 8 : 0)) & (B_ROWS - 1)];

  float rowSum[2][8] = {{0.f}};
  float rowPos[2][8] = {{0.f}};

  // cooperative tile loader: 256 threads x 32B == 16 cols x 512B
  const int ldc    = threadIdx.x >> 4;   // column within tile
  const int ldchnk = threadIdx.x & 15;   // 16-element chunk within column

  // prologue: stage tile 0
  {
    const __bf16* g = zn + (size_t)(n0base + ldc) * DDIM + ldchnk * 16;
    __bf16* l = &Btile[0][ldc * LDS_STRIDE + ldchnk * 16];
    async_copy16B(g, l);
    async_copy16B(g + 8, l + 8);
  }
  async_wait_all();
  __syncthreads();

  const int o2 = half ? 8 : 0;           // B-frag K offset (ISA 16-bit B layout)
  int buf = 0;

  for (int t = 0; t < TILES_PER_BLOCK; ++t) {
    // stage next tile into the other buffer while computing this one
    if (t + 1 < TILES_PER_BLOCK) {
      const __bf16* g = zn + (size_t)(n0base + (t + 1) * 16 + ldc) * DDIM + ldchnk * 16;
      __bf16* l = &Btile[buf ^ 1][ldc * LDS_STRIDE + ldchnk * 16];
      async_copy16B(g, l);
      async_copy16B(g + 8, l + 8);
    }

    const __bf16* bb = &Btile[buf][(size_t)l16 * LDS_STRIDE + o2];
    v8f C0 = {};
    v8f C1 = {};
#pragma unroll
    for (int k = 0; k < 8; ++k) {
      v8bf blo = *(const v8bf*)(bb + k * 32);
      v8bf bhi = *(const v8bf*)(bb + k * 32 + 16);
      v16bf Bf = __builtin_shufflevector(blo, bhi,
                   0, 1, 2, 3, 4, 5, 6, 7, 8, 9, 10, 11, 12, 13, 14, 15);
      C0 = __builtin_amdgcn_wmma_f32_16x16x32_bf16(false, Af[0][k], false, Bf,
                                                   (short)0, C0, false, false);
      C1 = __builtin_amdgcn_wmma_f32_16x16x32_bf16(false, Af[1][k], false, Bf,
                                                   (short)0, C1, false, false);
    }

    const int j    = n0base + t * 16 + l16;      // global column of this lane
    const int labJ = lab[j & (B_ROWS - 1)];
#pragma unroll
    for (int r = 0; r < 8; ++r) {
      {
        const int   i = iw0 + r + (half ? 8 : 0);
        const float v = C0[r] * TEMP_INV;
        if (labJ != labI[0][r]) rowSum[0][r] += __expf(v);
        if (j == (i ^ B_ROWS)) rowPos[0][r] += v;
      }
      {
        const int   i = iw0 + 16 + r + (half ? 8 : 0);
        const float v = C1[r] * TEMP_INV;
        if (labJ != labI[1][r]) rowSum[1][r] += __expf(v);
        if (j == (i ^ B_ROWS)) rowPos[1][r] += v;
      }
    }

    async_wait_all();
    __syncthreads();
    buf ^= 1;
  }

  // reduce across the 16-lane half owning each row, then write partials
#pragma unroll
  for (int s = 0; s < 2; ++s)
#pragma unroll
    for (int r = 0; r < 8; ++r) {
      float sv = rowSum[s][r], pv = rowPos[s][r];
#pragma unroll
      for (int m = 8; m >= 1; m >>= 1) {
        sv += __shfl_xor(sv, m, 32);
        pv += __shfl_xor(pv, m, 32);
      }
      rowSum[s][r] = sv; rowPos[s][r] = pv;
    }

  if (l16 == 0) {
#pragma unroll
    for (int s = 0; s < 2; ++s)
#pragma unroll
      for (int r = 0; r < 8; ++r) {
        const int row = iw0 + s * 16 + r + (half ? 8 : 0);
        partSum[(size_t)cs * NROWS + row] = rowSum[s][r];
        partPos[(size_t)cs * NROWS + row] = rowPos[s][r];
      }
  }
}

// ---------------------------------------------------------------------------
// Phase 3: combine the 8 column-split partials per row, deterministic
// single-block reduction to the scalar loss.
// ---------------------------------------------------------------------------
__global__ __launch_bounds__(256) void nscl_finish(const float* __restrict__ partSum,
                                                   const float* __restrict__ partPos,
                                                   float* __restrict__ out) {
  __shared__ float sbuf[8];
  float acc = 0.f;
  for (int i = threadIdx.x; i < NROWS; i += 256) {
    float s = 0.f, p = 0.f;
#pragma unroll
    for (int c = 0; c < NSPLIT; ++c) {
      s += partSum[(size_t)c * NROWS + i];
      p += partPos[(size_t)c * NROWS + i];
    }
    acc += logf(s + __expf(p)) - p;    // exp(pos) + sum_neg exp(sim)
  }
#pragma unroll
  for (int m = 16; m >= 1; m >>= 1) acc += __shfl_xor(acc, m, 32);
  if ((threadIdx.x & 31) == 0) sbuf[threadIdx.x >> 5] = acc;
  __syncthreads();
  if (threadIdx.x == 0) {
    float t = 0.f;
#pragma unroll
    for (int w = 0; w < 8; ++w) t += sbuf[w];
    out[0] = t / (float)NROWS;
  }
}

extern "C" void kernel_launch(void* const* d_in, const int* in_sizes, int n_in,
                              void* d_out, int out_size, void* d_ws, size_t ws_size,
                              hipStream_t stream) {
  (void)in_sizes; (void)n_in; (void)out_size; (void)ws_size;
  const float* zi  = (const float*)d_in[0];
  const float* zj  = (const float*)d_in[1];
  const int*   lab = (const int*)d_in[2];

  __bf16* zn      = (__bf16*)d_ws;                                     // 4 MB
  char*   wsb     = (char*)d_ws + (size_t)NROWS * DDIM * sizeof(__bf16);
  float*  partSum = (float*)wsb;                                       // 256 KB
  float*  partPos = (float*)(wsb + (size_t)NSPLIT * NROWS * sizeof(float));
  float*  out     = (float*)d_out;

  nscl_normalize<<<NROWS / 8, 256, 0, stream>>>(zi, zj, zn);
  nscl_simloss  <<<32 * NSPLIT, 256, 0, stream>>>(zn, lab, partSum, partPos);
  nscl_finish   <<<1, 256, 0, stream>>>(partSum, partPos, out);
}
